// Scaled_Dot_Product_Attention_20066087207356
// MI455X (gfx1250) — matrix operand order repaired and checked by reading, not compiled, not run on hardware
//
#include <hip/hip_runtime.h>

// Scaled dot-product attention (fwd, returns out AND full attn matrix).
// B=4, H=16, S=2048, D=64, fp32 in/out. Internally f16 WMMA, f32 accumulate,
// f32 softmax stats. One wave32 per 16-row query tile; 128KB LDS score buffer.

typedef __attribute__((ext_vector_type(16))) _Float16 v16h;
typedef __attribute__((ext_vector_type(8)))  float    v8f;
typedef __attribute__((ext_vector_type(4)))  float    f32x4;

#define SEQ   2048
#define HD    64
#define NKT   (SEQ / 16)      // 128 k-tiles per row block
#define QSCALE 0.125f         // 1/sqrt(64)

__global__ __launch_bounds__(32)
void sdpa_wmma_f16_kernel(const float* __restrict__ Q,
                          const float* __restrict__ K,
                          const float* __restrict__ V,
                          float* __restrict__ Out,
                          float* __restrict__ Attn)
{
    extern __shared__ float sS[];           // [16][SEQ] f32 scores / exp values

    const int lane = threadIdx.x;           // 0..31 (wave32)
    const int tile = blockIdx.x;            // bh * 128 + qt
    const int qt   = tile & (NKT - 1);
    const int bh   = tile >> 7;
    const int q0   = qt * 16;

    const float* Qb = Q + (size_t)bh * SEQ * HD;
    const float* Kb = K + (size_t)bh * SEQ * HD;
    const float* Vb = V + (size_t)bh * SEQ * HD;

    const int row16 = lane & 15;            // M (A/C row) / N (C col) index
    const int hi    = lane >> 4;            // half-wave select
    const int abase = hi ? 8 : 0;           // A-fragment K base per ISA layout

    // ---- Build Q A-fragments: 16x32 f16 layout, two K-chunks (D=64) ----
    // Lane L holds row M=L&15; K values: {abase+0..7, abase+16..23} per chunk.
    v16h aQ0, aQ1;
    {
        const float* qrow = Qb + (size_t)(q0 + row16) * HD;
        #pragma unroll
        for (int i = 0; i < 16; ++i) {
            int kk = abase + (i < 8 ? i : i + 8);
            aQ0[i] = (_Float16)(qrow[kk]      * QSCALE);
            aQ1[i] = (_Float16)(qrow[kk + 32] * QSCALE);
        }
    }

    // ---- Phase 1: scores(16 x 2048) = (Q*scale) @ K^T, staged in LDS ----
    for (int j = 0; j < NKT; ++j) {
        const int r0 = j * 16;
        // Prefetch next K tile into the cache hierarchy.
        if (j + 1 < NKT)
            __builtin_prefetch(Kb + (size_t)(r0 + 16) * HD + lane * 4, 0, 1);

        v8f sc = {};
        #pragma unroll
        for (int c = 0; c < 2; ++c) {
            // B = K^T chunk (32x16): lane holds N=row16, elements hold
            // K-dim {abase+0..7, abase+16..23} -> K[r0+n][c*32+k].
            v16h b;
            const float* kcol = Kb + (size_t)(r0 + row16) * HD + c * 32 + abase;
            #pragma unroll
            for (int i = 0; i < 16; ++i)
                b[i] = (_Float16)(kcol[i < 8 ? i : i + 8]);
            sc = __builtin_amdgcn_wmma_f32_16x16x32_f16(
                    false, (c == 0 ? aQ0 : aQ1), false, b,
                    (short)0, sc, false, false);
        }
        // C layout: VGPR v -> M = v + 8*hi, lane -> N = lane&15.
        const int mb = hi * 8;
        #pragma unroll
        for (int v = 0; v < 8; ++v)
            sS[(mb + v) * SEQ + r0 + row16] = sc[v];
    }
    __syncthreads();

    // ---- Phase 2: softmax stats (2 lanes per row, halves combined) ----
    float* myhalf = sS + row16 * SEQ + hi * (SEQ / 2);
    float m = -3.402823466e38f;
    for (int i = 0; i < SEQ / 2; ++i) m = fmaxf(m, myhalf[i]);
    m = fmaxf(m, __shfl_xor(m, 16, 32));

    float s = 0.0f;
    for (int i = 0; i < SEQ / 2; ++i) {
        float e = __expf(myhalf[i] - m);
        myhalf[i] = e;                      // keep unnormalized exp in LDS
        s += e;
    }
    s += __shfl_xor(s, 16, 32);
    const float inv = 1.0f / s;
    __syncthreads();

    // ---- Phase 3: stream attn tile to global (contiguous 128KB, f32x4) ----
    {
        float* dst = Attn + (size_t)bh * SEQ * SEQ + (size_t)q0 * SEQ;
        #pragma unroll 4
        for (int t = 0; t < (16 * SEQ) / (32 * 4); ++t) {   // 256 iters
            const int flat = t * 128 + lane * 4;
            const float invr = __shfl(inv, t >> 4, 32);     // row = t/16
            f32x4 p = *(const f32x4*)(sS + flat);
            p *= invr;
            *(f32x4*)(dst + flat) = p;
        }
    }

    // ---- Phase 4: out(16x64) = P @ V, 64 K-chunks of 32, 4 N-tiles ----
    v8f acc0 = {}, acc1 = {}, acc2 = {}, acc3 = {};
    for (int c = 0; c < SEQ / 32; ++c) {
        // A = exp-P chunk from LDS (unnormalized, in (0,1], f16-friendly).
        v16h aP;
        const float* prow = sS + row16 * SEQ + c * 32;
        #pragma unroll
        for (int i = 0; i < 16; ++i)
            aP[i] = (_Float16)(prow[abase + (i < 8 ? i : i + 8)]);

        // B tiles: lane holds N=row16; elements hold K-dim rows (same order as A).
        const float* vrow = Vb + (size_t)(c * 32) * HD + row16;
        v16h b0, b1, b2, b3;
        #pragma unroll
        for (int i = 0; i < 16; ++i) {
            const int kr = (abase + (i < 8 ? i : i + 8)) * HD;
            b0[i] = (_Float16)vrow[kr];
            b1[i] = (_Float16)vrow[kr + 16];
            b2[i] = (_Float16)vrow[kr + 32];
            b3[i] = (_Float16)vrow[kr + 48];
        }
        acc0 = __builtin_amdgcn_wmma_f32_16x16x32_f16(false, aP, false, b0,
                                                      (short)0, acc0, false, false);
        acc1 = __builtin_amdgcn_wmma_f32_16x16x32_f16(false, aP, false, b1,
                                                      (short)0, acc1, false, false);
        acc2 = __builtin_amdgcn_wmma_f32_16x16x32_f16(false, aP, false, b2,
                                                      (short)0, acc2, false, false);
        acc3 = __builtin_amdgcn_wmma_f32_16x16x32_f16(false, aP, false, b3,
                                                      (short)0, acc3, false, false);
    }

    // ---- Phase 5: normalize rows (1/sum via shfl) and store out ----
    float* orow = Out + ((size_t)bh * SEQ + q0) * HD;
    #pragma unroll
    for (int v = 0; v < 8; ++v) {
        const int mrow = hi * 8 + v;
        const float invr = __shfl(inv, mrow, 32);   // lane r holds row r's inv
        orow[(size_t)mrow * HD +  0 + row16] = acc0[v] * invr;
        orow[(size_t)mrow * HD + 16 + row16] = acc1[v] * invr;
        orow[(size_t)mrow * HD + 32 + row16] = acc2[v] * invr;
        orow[(size_t)mrow * HD + 48 + row16] = acc3[v] * invr;
    }
}

extern "C" void kernel_launch(void* const* d_in, const int* in_sizes, int n_in,
                              void* d_out, int out_size, void* d_ws, size_t ws_size,
                              hipStream_t stream) {
    (void)in_sizes; (void)n_in; (void)d_ws; (void)ws_size; (void)out_size;
    const float* Q = (const float*)d_in[0];
    const float* K = (const float*)d_in[1];
    const float* V = (const float*)d_in[2];

    const int Bn = 4, Hn = 16;
    float* out  = (float*)d_out;                                  // [B,H,S,D]
    float* attn = out + (size_t)Bn * Hn * SEQ * HD;               // [B,H,S,S]

    dim3 grid(Bn * Hn * (SEQ / 16));   // 8192 single-wave workgroups
    dim3 block(32);
    size_t lds_bytes = (size_t)16 * SEQ * sizeof(float);          // 128 KB
    sdpa_wmma_f16_kernel<<<grid, block, lds_bytes, stream>>>(Q, K, V, out, attn);
}
